// Attention_47605417509124
// MI455X (gfx1250) — matrix operand order repaired and checked by reading, not compiled, not run on hardware
//
#include <hip/hip_runtime.h>
#include <hip/hip_bf16.h>
#include <stdint.h>

#define BATCH 4
#define NSEQ  4096
#define CDIM  256
#define BNROWS (BATCH * NSEQ)
#define WMAT_ELEMS (CDIM * CDIM)
#define NTILES (NSEQ / 32)

typedef __bf16 bf16;
typedef __attribute__((ext_vector_type(16))) __bf16 v16bf;
typedef __attribute__((ext_vector_type(8)))  float  v8f;

union FragU { v16bf v; uint32_t u[8]; };
union PkU   { __bf16 h[2]; uint32_t u; };

// ---- WMMA wrapper: D = A(16x32 bf16) * B(32x16 bf16) + C(16x16 f32) ----
static __device__ inline v8f wmma_bf16(v16bf a, v16bf b, v8f c) {
  return __builtin_amdgcn_wmma_f32_16x16x32_bf16(
      /*neg_a=*/false, a, /*neg_b=*/false, b,
      /*c_mod=*/(short)0, c, /*reuse_a=*/false, /*reuse_b=*/false);
}

// ---- A-fragment (16x32, rows = M) from row-major bf16 source ----
// lane(0-15)=M, half h=lane>>4; VGPR v holds K = (v>>2)*16 + h*8 + (v&3)*2, +1
static __device__ inline v16bf ldfrag_a_bf(const bf16* base, int stride, int row0, int col0) {
  int lane = threadIdx.x & 31;
  int r = row0 + (lane & 15);
  int h = (lane >> 4) & 1;
  const bf16* rp = base + (size_t)r * stride + col0;
  FragU f;
#pragma unroll
  for (int v = 0; v < 8; ++v) {
    int k = ((v >> 2) << 4) + (h << 3) + ((v & 3) << 1);
    f.u[v] = *(const uint32_t*)(rp + k);
  }
  return f.v;
}

static __device__ inline v16bf ldfrag_a_f32(const float* base, int stride, int row0, int col0) {
  int lane = threadIdx.x & 31;
  int r = row0 + (lane & 15);
  int h = (lane >> 4) & 1;
  const float* rp = base + (size_t)r * stride + col0;
  FragU f;
#pragma unroll
  for (int v = 0; v < 8; ++v) {
    int k = ((v >> 2) << 4) + (h << 3) + ((v & 3) << 1);
    PkU p; p.h[0] = (bf16)rp[k]; p.h[1] = (bf16)rp[k + 1];
    f.u[v] = p.u;
  }
  return f.v;
}

// ---- B-fragment (32x16, B[k][n] = src[n][k], src row-major) ----
// lane(0-15)=N, half h; VGPR v holds K = (v>>2)*16 + h*8 + (v&3)*2, +1
static __device__ inline v16bf ldfrag_b_bf(const bf16* base, int stride, int row0, int col0) {
  int lane = threadIdx.x & 31;
  int r = row0 + (lane & 15);
  int h = (lane >> 4) & 1;
  const bf16* rp = base + (size_t)r * stride + col0 + (h << 3);
  FragU f;
#pragma unroll
  for (int v = 0; v < 8; ++v)
    f.u[v] = *(const uint32_t*)(rp + ((v >> 2) << 4) + ((v & 3) << 1));
  return f.v;
}

// =====================================================================
// Kernel 0: one-shot fp32 -> bf16 conversion of the 4 weight matrices
// =====================================================================
__global__ __launch_bounds__(256) void wcvt_kernel(
    const float* __restrict__ gw, const float* __restrict__ tw,
    const float* __restrict__ pw, const float* __restrict__ ww,
    bf16* __restrict__ wbf) {
  int i = (blockIdx.x * 256 + threadIdx.x) * 4;  // 64 blocks cover 65536 elems
  union { bf16 e[4]; uint2 q; } o;
#define CVT4(src, slot)                                                     \
  {                                                                         \
    float4 f = *(const float4*)((src) + i);                                 \
    o.e[0] = (bf16)f.x; o.e[1] = (bf16)f.y;                                 \
    o.e[2] = (bf16)f.z; o.e[3] = (bf16)f.w;                                 \
    *(uint2*)(wbf + (size_t)(slot) * WMAT_ELEMS + i) = o.q;                 \
  }
  CVT4(gw, 0)
  CVT4(tw, 1)
  CVT4(pw, 2)
  CVT4(ww, 3)
#undef CVT4
}

// ---- shared GEMM tile: out[16x256] = A(16x256) * W^T + bias, bf16 out ----
static __device__ inline void lin_tile_store_bf16(
    const v16bf* xa, const bf16* __restrict__ wb, const float* __restrict__ bv,
    bf16* __restrict__ ov, int row0) {
  int lane = threadIdx.x & 31;
  int n = lane & 15, hh = (lane >> 4) & 1;
  for (int oc = 0; oc < 16; ++oc) {
    v8f acc = {0.f, 0.f, 0.f, 0.f, 0.f, 0.f, 0.f, 0.f};
#pragma unroll
    for (int kc = 0; kc < 8; ++kc) {
      v16bf bf = ldfrag_b_bf(wb, CDIM, oc * 16, kc * 32);  // B[k][n] = w[d][k]
      acc = wmma_bf16(xa[kc], bf, acc);
    }
    int d = oc * 16 + n;
    float bias = bv[d];
#pragma unroll
    for (int v = 0; v < 8; ++v) {
      int m = hh * 8 + v;
      ov[(size_t)(row0 + m) * CDIM + d] = (bf16)(acc[v] + bias);
    }
  }
}

// =====================================================================
// Kernel 1: g_x / theta_x / phi_x projections -> bf16 workspace
// =====================================================================
__global__ __launch_bounds__(256) void proj_kernel(
    const float* __restrict__ x, const bf16* __restrict__ wbf,
    const float* __restrict__ gb, const float* __restrict__ tb,
    const float* __restrict__ pb,
    bf16* __restrict__ og, bf16* __restrict__ ot, bf16* __restrict__ op) {
  int w    = threadIdx.x >> 5;
  int row0 = blockIdx.x * 128 + w * 16;

  v16bf xa[8];
#pragma unroll
  for (int kc = 0; kc < 8; ++kc) xa[kc] = ldfrag_a_f32(x, CDIM, row0, kc * 32);

  lin_tile_store_bf16(xa, wbf + 0 * WMAT_ELEMS, gb, og, row0);
  lin_tile_store_bf16(xa, wbf + 1 * WMAT_ELEMS, tb, ot, row0);
  lin_tile_store_bf16(xa, wbf + 2 * WMAT_ELEMS, pb, op, row0);
}

// =====================================================================
// Kernel 2: y = (phi @ theta^T / N) @ g   per batch
// grid (32, 4), block 256 (8 waves); wave owns 16 rows x 256 cols of y.
// theta tiles: async-to-LDS, ping-pong double buffer (ASYNCcnt pipelined)
// g tiles:     sync loads + manual transpose into LDS
// =====================================================================
__global__ __launch_bounds__(256) void attn_kernel(
    const bf16* __restrict__ ph, const bf16* __restrict__ th,
    const bf16* __restrict__ gx, bf16* __restrict__ y) {
  __shared__ bf16 sTh[2][32][264];   // theta tile ping-pong (async dest)
  __shared__ bf16 sGT[256][36];      // g tile transposed [c][j]
  __shared__ bf16 sS[8][16][34];     // per-wave S scratch (16x32)

  int b    = blockIdx.y;
  int t    = threadIdx.x;
  int w    = t >> 5;
  int lane = t & 31;
  int n = lane & 15, hh = (lane >> 4) & 1;
  int row0 = blockIdx.x * 128 + w * 16;

  const bf16* phB = ph + (size_t)b * NSEQ * CDIM;
  const bf16* thB = th + (size_t)b * NSEQ * CDIM;
  const bf16* gB  = gx + (size_t)b * NSEQ * CDIM;

  v16bf pa[8];
#pragma unroll
  for (int kc = 0; kc < 8; ++kc) pa[kc] = ldfrag_a_bf(phB, CDIM, row0, kc * 32);

  v8f acc[16];
#pragma unroll
  for (int i = 0; i < 16; ++i) acc[i] = (v8f){0.f,0.f,0.f,0.f,0.f,0.f,0.f,0.f};

  const float scale = 1.0f / (float)NSEQ;
  int jrow = t >> 3, c0 = (t & 7) << 5;   // per-thread staging coords

  // issue 4 async b128 loads (this thread's share of a 32x256 theta tile)
  auto stage_theta_async = [&](int buf, int j0) {
#pragma unroll
    for (int i = 0; i < 4; ++i) {
      int cid = t * 4 + i;                  // 0..1023 chunks of 8 bf16
      int r = cid >> 5, c = (cid & 31) << 3;
      uint32_t lo = (uint32_t)(size_t)&sTh[buf][r][c];
      uint64_t ga = (uint64_t)(size_t)(thB + (size_t)(j0 + r) * CDIM + c);
      asm volatile("global_load_async_to_lds_b128 %0, %1, off"
                   :: "v"(lo), "v"(ga) : "memory");
    }
  };

  stage_theta_async(0, 0);   // prestage tile 0

  for (int jt = 0; jt < NTILES; ++jt) {
    int j0  = jt * 32;
    int cur = jt & 1;
    __syncthreads();  // B1: all waves done computing previous tile
    // pipeline: next theta tile -> other buffer while we compute this one
    if (jt + 1 < NTILES) stage_theta_async(cur ^ 1, j0 + 32);

    // --- g tile: sync load + transpose into sGT ---
    union { uint4 q[4]; bf16 e[32]; } tmp;
    const bf16* gp = gB + (size_t)(j0 + jrow) * CDIM + c0;
#pragma unroll
    for (int i = 0; i < 4; ++i) tmp.q[i] = ((const uint4*)gp)[i];
    if (jt + 1 < NTILES) {
      __builtin_prefetch(gB + (size_t)(j0 + 32 + jrow) * CDIM + c0, 0, 1);
      // async completes in order: <=4 outstanding => this tile's 4 are done,
      // next tile's 4 remain in flight
      asm volatile("s_wait_asynccnt 0x4" ::: "memory");
    } else {
      asm volatile("s_wait_asynccnt 0x0" ::: "memory");
    }
#pragma unroll
    for (int i = 0; i < 32; ++i) sGT[c0 + i][jrow] = tmp.e[i];
    __syncthreads();  // B2: theta[cur] + gT visible to all waves

    // --- S = phi_tile(16x256) @ theta_chunk^T(256x32) : 16 WMMAs ---
    v8f s0 = {0.f,0.f,0.f,0.f,0.f,0.f,0.f,0.f};
    v8f s1 = {0.f,0.f,0.f,0.f,0.f,0.f,0.f,0.f};
#pragma unroll
    for (int kc = 0; kc < 8; ++kc) {
      v16bf b0 = ldfrag_b_bf(&sTh[cur][0][0], 264, 0,  kc * 32);
      s0 = wmma_bf16(pa[kc], b0, s0);
      v16bf b1 = ldfrag_b_bf(&sTh[cur][0][0], 264, 16, kc * 32);
      s1 = wmma_bf16(pa[kc], b1, s1);
    }
    // --- S (f32, C/D layout) -> bf16 in LDS, folding 1/N ---
    // per-wave scratch + same-wave LDS ordering: no block barrier needed
#pragma unroll
    for (int v = 0; v < 8; ++v) {
      int m = hh * 8 + v;
      sS[w][m][n]      = (bf16)(s0[v] * scale);
      sS[w][m][16 + n] = (bf16)(s1[v] * scale);
    }
    v16bf sa = ldfrag_a_bf(&sS[w][0][0], 34, 0, 0);
    // --- y(16x256) += S(16x32) @ g_chunk(32x256) : 16 WMMAs ---
#pragma unroll
    for (int cc = 0; cc < 16; ++cc) {
      v16bf bg = ldfrag_b_bf(&sGT[0][0], 36, cc * 16, 0);
      acc[cc] = wmma_bf16(sa, bg, acc[cc]);
    }
  }

  // --- write y as bf16 ---
#pragma unroll
  for (int cc = 0; cc < 16; ++cc) {
#pragma unroll
    for (int v = 0; v < 8; ++v) {
      int m = hh * 8 + v;
      y[((size_t)b * NSEQ + row0 + m) * CDIM + cc * 16 + n] = (bf16)acc[cc][v];
    }
  }
}

// =====================================================================
// Kernel 3: out = y @ W^T + W_b + x   (fp32 out)
// =====================================================================
__global__ __launch_bounds__(256) void final_kernel(
    const bf16* __restrict__ yb, const bf16* __restrict__ wbf,
    const float* __restrict__ wb, const float* __restrict__ x,
    float* __restrict__ out) {
  int w    = threadIdx.x >> 5;
  int row0 = blockIdx.x * 128 + w * 16;
  int lane = threadIdx.x & 31;
  int n = lane & 15, hh = (lane >> 4) & 1;
  const bf16* wm = wbf + 3 * WMAT_ELEMS;

  v16bf ya[8];
#pragma unroll
  for (int kc = 0; kc < 8; ++kc) ya[kc] = ldfrag_a_bf(yb, CDIM, row0, kc * 32);

  for (int oc = 0; oc < 16; ++oc) {
    v8f acc = {0.f, 0.f, 0.f, 0.f, 0.f, 0.f, 0.f, 0.f};
#pragma unroll
    for (int kc = 0; kc < 8; ++kc) {
      v16bf bf = ldfrag_b_bf(wm, CDIM, oc * 16, kc * 32);
      acc = wmma_bf16(ya[kc], bf, acc);
    }
    int d = oc * 16 + n;
    float bias = wb[d];
#pragma unroll
    for (int v = 0; v < 8; ++v) {
      int m = hh * 8 + v;
      size_t idx = (size_t)(row0 + m) * CDIM + d;
      out[idx] = acc[v] + bias + x[idx];
    }
  }
}

// =====================================================================
extern "C" void kernel_launch(void* const* d_in, const int* in_sizes, int n_in,
                              void* d_out, int out_size, void* d_ws, size_t ws_size,
                              hipStream_t stream) {
  (void)in_sizes; (void)n_in; (void)out_size; (void)ws_size;
  const float* x   = (const float*)d_in[0];
  const float* g_w = (const float*)d_in[1];
  const float* g_b = (const float*)d_in[2];
  const float* t_w = (const float*)d_in[3];
  const float* t_b = (const float*)d_in[4];
  const float* p_w = (const float*)d_in[5];
  const float* p_b = (const float*)d_in[6];
  const float* W_w = (const float*)d_in[7];
  const float* W_b = (const float*)d_in[8];
  float* out = (float*)d_out;

  // workspace: 4 bf16 activation buffers [BN, C] (8 MB each) + 4 bf16 weights
  bf16* g_bf  = (bf16*)d_ws;
  bf16* th_bf = g_bf  + (size_t)BNROWS * CDIM;
  bf16* ph_bf = th_bf + (size_t)BNROWS * CDIM;
  bf16* y_bf  = ph_bf + (size_t)BNROWS * CDIM;
  bf16* w_bf  = y_bf  + (size_t)BNROWS * CDIM;   // 4 * 64K bf16 = 512 KB

  wcvt_kernel<<<WMAT_ELEMS / (256 * 4), 256, 0, stream>>>(g_w, t_w, p_w, W_w, w_bf);
  proj_kernel<<<BNROWS / 128, 256, 0, stream>>>(x, w_bf, g_b, t_b, p_b,
                                                g_bf, th_bf, ph_bf);
  attn_kernel<<<dim3(NSEQ / 128, BATCH), 256, 0, stream>>>(ph_bf, th_bf, g_bf, y_bf);
  final_kernel<<<BNROWS / 128, 256, 0, stream>>>(y_bf, w_bf, W_b, x, out);
}
